// MRGCN_4578435137887
// MI455X (gfx1250) — matrix operand order repaired and checked by reading, not compiled, not run on hardware
//
#include <hip/hip_runtime.h>
#include <cstdint>

// ---------------- problem constants (match reference) ----------------
#define NN      50000          // nodes
#define NE      800000         // edges
#define FEAT    128
#define NREL    8
#define NBASES  30
#define NCLASS  8
#define NGRAPH  64
#define M_PAD   50048          // 391 * 128, padded M for full-EXEC WMMA tiles
#define KA      1024           // 8 relations * 128
#define KTOT    1152           // KA + 128 (root term)
#define NKSTEP  36             // KTOT / 32

typedef __attribute__((ext_vector_type(16))) __bf16 bf16x16;
typedef __attribute__((ext_vector_type(8)))  float  f32x8;

// Fragment buffer: [kstep(36)][osub(8)][lane(32)][elem(16)] bf16
#define FRAG_ELEMS (NKSTEP * 8 * 32 * 16)

// ---------------------------------------------------------------------
// Build combined weight W[k][o] (k<1024: relation part from basis/comp,
// k>=1024: root) as bf16 hi/lo WMMA B-fragments, pre-swizzled so the GEMM
// does a single coalesced 32B load per fragment per lane.
// B layout (v_wmma_*_16x16x32 B operand): lane = n + 16*((k>>3)&1),
// element e = (k&7) + 8*(k>>4) within 32-chunk (table order, matches A).
__global__ void mrgcn_build_bfrag(const float* __restrict__ basis,
                                  const float* __restrict__ comp,
                                  const float* __restrict__ root,
                                  __bf16* __restrict__ bhi,
                                  __bf16* __restrict__ blo) {
    int idx = blockIdx.x * blockDim.x + threadIdx.x;
    if (idx >= KTOT * FEAT) return;
    int k = idx / FEAT;
    int o = idx % FEAT;
    float val;
    if (k < KA) {
        int r = k >> 7, i = k & 127;
        float s = 0.f;
#pragma unroll
        for (int b = 0; b < NBASES; ++b)
            s += comp[r * NBASES + b] * basis[((size_t)b * FEAT + i) * FEAT + o];
        val = s;
    } else {
        val = root[(size_t)(k - KA) * FEAT + o];
    }
    __bf16 h = (__bf16)val;
    __bf16 l = (__bf16)(val - (float)h);
    int kstep = k >> 5, kk = k & 31;
    int osub = o >> 4;
    size_t fi = (((size_t)(kstep * 8 + osub)) * 32 + (o & 15) + 16 * ((kk >> 3) & 1)) * 16 + (kk & 7) + 8 * (kk >> 4);
    bhi[fi] = h;
    blo[fi] = l;
}

// ---------------- degree -------------------------------------------------
__global__ void mrgcn_deg_scatter(const int* __restrict__ ei, float* __restrict__ deg) {
    int e = blockIdx.x * blockDim.x + threadIdx.x;
    if (e >= NE) return;
    int dst = ei[NE + e];
    atomicAdd(&deg[dst], 1.0f);
}

__global__ void mrgcn_invdeg(const float* __restrict__ deg, float* __restrict__ invdeg) {
    int n = blockIdx.x * blockDim.x + threadIdx.x;
    if (n >= M_PAD) return;
    invdeg[n] = 1.0f / fmaxf(deg[n], 1.0f);
}

// ---------------- per-edge scatter: Aop[dst][r*128 + i] += x[src][i] -----
// One wave (32 lanes) per edge; each lane handles 4 consecutive features.
__global__ void mrgcn_edge_scatter(const float* __restrict__ xin,
                                   const int* __restrict__ ei,
                                   const int* __restrict__ attr,
                                   float* __restrict__ Aop) {
    int tid  = blockIdx.x * blockDim.x + threadIdx.x;
    int e    = tid >> 5;
    int lane = tid & 31;
    if (e >= NE) return;
    int src = ei[e];
    int dst = ei[NE + e];
    int r   = attr[e];
    float4 xv = *(const float4*)(xin + (size_t)src * FEAT + lane * 4);
    float* p  = Aop + (size_t)dst * KA + (size_t)r * FEAT + lane * 4;
    atomicAdd(p + 0, xv.x);
    atomicAdd(p + 1, xv.y);
    atomicAdd(p + 2, xv.z);
    atomicAdd(p + 3, xv.w);
}

// ---------------- fused RGCN GEMM ---------------------------------------
// out[n][o] = act( invdeg[n]*sum_{k<1024} Aop[n][k]*W[k][o]
//                + sum_{k<128} x[n][k]*root[k][o] + bias[o] )
// 256 threads = 8 waves; each wave owns a 16(M)x128(O) stripe (8 subtiles).
// K loop in 32-chunks; f32 operands split hi/lo bf16 -> 3 WMMAs per subtile
// (a_hi*b_hi + a_lo*b_hi + a_hi*b_lo), f32 accumulate: ~f32 accuracy.
__global__ __launch_bounds__(256)
void mrgcn_gemm(const float* __restrict__ Aop,
                const float* __restrict__ xin,
                const float* __restrict__ invdeg,
                const __bf16* __restrict__ Bhi,
                const __bf16* __restrict__ Blo,
                const float* __restrict__ bias,
                float* __restrict__ out,
                int do_relu) {
    int wave = threadIdx.x >> 5;
    int lane = threadIdx.x & 31;
    int m0   = blockIdx.x * 128 + wave * 16;
    int mrow = m0 + (lane & 15);
    int half = lane >> 4;

    const float scaleA = invdeg[mrow];
    const float* rowA  = Aop + (size_t)mrow * KA;
    const float* rowX  = xin + (size_t)mrow * FEAT;

    f32x8 acc[8];
#pragma unroll
    for (int s = 0; s < 8; ++s) acc[s] = (f32x8)(0.0f);

    for (int kstep = 0; kstep < NKSTEP; ++kstep) {
        const float* rowp;
        int   k0;
        float sc;
        if (kstep < 32) { rowp = rowA; k0 = kstep * 32;        sc = scaleA; }
        else            { rowp = rowX; k0 = (kstep - 32) * 32; sc = 1.0f;   }

        // A fragment: lane(&15)=row, half selects K runs {0..7,16..23} / {8..15,24..31}
        float4 f0 = *(const float4*)(rowp + k0 + 8 * half);
        float4 f1 = *(const float4*)(rowp + k0 + 8 * half + 4);
        float4 f2 = *(const float4*)(rowp + k0 + 16 + 8 * half);
        float4 f3 = *(const float4*)(rowp + k0 + 16 + 8 * half + 4);
        float af[16];
        af[0]=f0.x;  af[1]=f0.y;  af[2]=f0.z;  af[3]=f0.w;
        af[4]=f1.x;  af[5]=f1.y;  af[6]=f1.z;  af[7]=f1.w;
        af[8]=f2.x;  af[9]=f2.y;  af[10]=f2.z; af[11]=f2.w;
        af[12]=f3.x; af[13]=f3.y; af[14]=f3.z; af[15]=f3.w;

        bf16x16 ahi, alo;
#pragma unroll
        for (int i = 0; i < 16; ++i) {
            float a  = af[i] * sc;
            __bf16 h = (__bf16)a;
            ahi[i] = h;
            alo[i] = (__bf16)(a - (float)h);
        }

#pragma unroll
        for (int osub = 0; osub < 8; ++osub) {
            size_t bi  = (((size_t)(kstep * 8 + osub)) * 32 + lane) * 16;
            bf16x16 bh = *(const bf16x16*)(Bhi + bi);
            bf16x16 bl = *(const bf16x16*)(Blo + bi);
            acc[osub] = __builtin_amdgcn_wmma_f32_16x16x32_bf16(
                false, ahi, false, bh, (short)0, acc[osub], false, false);
            acc[osub] = __builtin_amdgcn_wmma_f32_16x16x32_bf16(
                false, alo, false, bh, (short)0, acc[osub], false, false);
            acc[osub] = __builtin_amdgcn_wmma_f32_16x16x32_bf16(
                false, ahi, false, bl, (short)0, acc[osub], false, false);
        }
    }

    // C/D layout: lane L, VGPR r -> row = r + 8*(L>>4), col = osub*16 + (L&15)
#pragma unroll
    for (int osub = 0; osub < 8; ++osub) {
        int col  = osub * 16 + (lane & 15);
        float bv = bias[col];
#pragma unroll
        for (int r = 0; r < 8; ++r) {
            int row  = m0 + r + 8 * half;
            float vv = acc[osub][r] + bv;
            if (do_relu) vv = fmaxf(vv, 0.0f);
            out[(size_t)row * FEAT + col] = vv;
        }
    }
}

// ---------------- pooling + head -----------------------------------------
__global__ void mrgcn_pool(const float* __restrict__ h, const int* __restrict__ batch,
                           float* __restrict__ gsum, float* __restrict__ gcnt) {
    int tid = blockIdx.x * blockDim.x + threadIdx.x;
    int n = tid >> 7, c = tid & 127;
    if (n >= NN) return;
    int g = batch[n];
    atomicAdd(&gsum[g * FEAT + c], h[(size_t)n * FEAT + c]);
    if (c == 0) atomicAdd(&gcnt[g], 1.0f);
}

__global__ void mrgcn_head(const float* __restrict__ gsum, const float* __restrict__ gcnt,
                           const float* __restrict__ fc1w, const float* __restrict__ fc1b,
                           const float* __restrict__ fc2w, const float* __restrict__ fc2b,
                           float* __restrict__ out) {
    __shared__ float v[FEAT];
    __shared__ float u[FEAT];
    __shared__ float lg[NCLASS];
    __shared__ float red[2];
    int t = threadIdx.x;
    float s = 0.f;
    for (int g = 0; g < NGRAPH; ++g)
        s += gsum[g * FEAT + t] / fmaxf(gcnt[g], 1.0f);
    v[t] = s * (1.0f / NGRAPH);
    __syncthreads();
    float a = fc1b[t];
    for (int c = 0; c < FEAT; ++c) a += v[c] * fc1w[c * FEAT + t];
    u[t] = fmaxf(a, 0.f);
    __syncthreads();
    if (t < NCLASS) {
        float l = fc2b[t];
        for (int o = 0; o < FEAT; ++o) l += u[o] * fc2w[o * NCLASS + t];
        lg[t] = l;
    }
    __syncthreads();
    if (t == 0) {
        float m = lg[0];
        for (int j = 1; j < NCLASS; ++j) m = fmaxf(m, lg[j]);
        float se = 0.f;
        for (int j = 0; j < NCLASS; ++j) se += __expf(lg[j] - m);
        red[0] = m;
        red[1] = __logf(se);
    }
    __syncthreads();
    if (t < NCLASS) out[t] = lg[t] - red[0] - red[1];
}

// ---------------------------------------------------------------------
extern "C" void kernel_launch(void* const* d_in, const int* in_sizes, int n_in,
                              void* d_out, int out_size, void* d_ws, size_t ws_size,
                              hipStream_t stream) {
    const float* x      = (const float*)d_in[0];
    const float* basis1 = (const float*)d_in[1];
    const float* comp1  = (const float*)d_in[2];
    const float* root1  = (const float*)d_in[3];
    const float* bias1  = (const float*)d_in[4];
    const float* basis2 = (const float*)d_in[5];
    const float* comp2  = (const float*)d_in[6];
    const float* root2  = (const float*)d_in[7];
    const float* bias2  = (const float*)d_in[8];
    const float* fc1w   = (const float*)d_in[9];
    const float* fc1b   = (const float*)d_in[10];
    const float* fc2w   = (const float*)d_in[11];
    const float* fc2b   = (const float*)d_in[12];
    const int*   eidx   = (const int*)d_in[13];   // integer inputs arrive as int32
    const int*   eattr  = (const int*)d_in[14];
    const int*   batch  = (const int*)d_in[15];
    float* out = (float*)d_out;

    // -------- workspace layout (~258 MB) --------
    uint8_t* ws = (uint8_t*)d_ws;
    size_t off = 0;
    auto alloc = [&](size_t bytes) -> void* {
        void* p = ws + off;
        off += (bytes + 255) & ~(size_t)255;
        return p;
    };
    float*  hA     = (float*)alloc((size_t)M_PAD * FEAT * 4);
    float*  hB     = (float*)alloc((size_t)M_PAD * FEAT * 4);
    float*  Aop    = (float*)alloc((size_t)M_PAD * KA * 4);
    float*  deg    = (float*)alloc((size_t)M_PAD * 4);
    float*  invdeg = (float*)alloc((size_t)M_PAD * 4);
    __bf16* B1h    = (__bf16*)alloc(FRAG_ELEMS * 2);
    __bf16* B1l    = (__bf16*)alloc(FRAG_ELEMS * 2);
    __bf16* B2h    = (__bf16*)alloc(FRAG_ELEMS * 2);
    __bf16* B2l    = (__bf16*)alloc(FRAG_ELEMS * 2);
    float*  gsum   = (float*)alloc((size_t)NGRAPH * FEAT * 4);
    float*  gcnt   = (float*)alloc((size_t)NGRAPH * 4);
    if (off > ws_size) return;  // workspace too small; bail deterministically

    // -------- weights -> pre-swizzled bf16 hi/lo B-fragments --------
    {
        int nthr = KTOT * FEAT;
        mrgcn_build_bfrag<<<(nthr + 255) / 256, 256, 0, stream>>>(basis1, comp1, root1, B1h, B1l);
        mrgcn_build_bfrag<<<(nthr + 255) / 256, 256, 0, stream>>>(basis2, comp2, root2, B2h, B2l);
    }

    // -------- degrees (shared across all 6 layers) --------
    hipMemsetAsync(deg, 0, (size_t)M_PAD * 4, stream);
    mrgcn_deg_scatter<<<(NE + 255) / 256, 256, 0, stream>>>(eidx, deg);
    mrgcn_invdeg<<<(M_PAD + 255) / 256, 256, 0, stream>>>(deg, invdeg);

    // -------- x -> padded ping buffer --------
    hipMemsetAsync(hA, 0, (size_t)M_PAD * FEAT * 4, stream);
    hipMemcpyAsync(hA, x, (size_t)NN * FEAT * 4, hipMemcpyDeviceToDevice, stream);

    // -------- 6 conv layers --------
    float* cur = hA;
    float* nxt = hB;
    for (int L = 0; L < 6; ++L) {
        const __bf16* bh   = (L == 0) ? B1h : B2h;
        const __bf16* bl   = (L == 0) ? B1l : B2l;
        const float*  bias = (L == 0) ? bias1 : bias2;
        int relu = (L == 1) ? 0 : 1;  // relu(conv1), conv2, then 4x relu(conv2)

        hipMemsetAsync(Aop, 0, (size_t)M_PAD * KA * 4, stream);
        mrgcn_edge_scatter<<<(NE * 32) / 256, 256, 0, stream>>>(cur, eidx, eattr, Aop);
        mrgcn_gemm<<<M_PAD / 128, 256, 0, stream>>>(Aop, cur, invdeg, bh, bl, bias, nxt, relu);

        float* t = cur; cur = nxt; nxt = t;
    }

    // -------- pool + head --------
    hipMemsetAsync(gsum, 0, (size_t)NGRAPH * FEAT * 4, stream);
    hipMemsetAsync(gcnt, 0, (size_t)NGRAPH * 4, stream);
    mrgcn_pool<<<((size_t)NN * FEAT + 255) / 256, 256, 0, stream>>>(cur, batch, gsum, gcnt);
    mrgcn_head<<<1, 128, 0, stream>>>(gsum, gcnt, fc1w, fc1b, fc2w, fc2b, out);
}
